// MultiHeadAttention_6150393168574
// MI455X (gfx1250) — compile-verified, matrix operand order checked by reading, not run on hardware
//
#include <hip/hip_runtime.h>

typedef __bf16 bf16;
typedef __attribute__((ext_vector_type(8)))  bf16  bf16x8;
typedef __attribute__((ext_vector_type(16))) bf16  v16bf;
typedef __attribute__((ext_vector_type(8)))  float v8f;
typedef __attribute__((ext_vector_type(4)))  float floatx4;

constexpr int SEQ    = 1024;
constexpr int DMODEL = 1024;
constexpr int NBATCH = 8;
constexpr int NHEAD  = 16;
constexpr int DHEAD  = 64;
constexpr int BS     = NBATCH * SEQ;   // 8192 rows

// ---------------------------------------------------------------------------
// WMMA helpers
// ---------------------------------------------------------------------------
static __device__ inline v8f wmma_bf16(v16bf a, v16bf b, v8f c) {
  // D = A(16x32 bf16) * B(32x16 bf16) + C(16x16 f32)
  return __builtin_amdgcn_wmma_f32_16x16x32_bf16(
      /*neg_a=*/false, a, /*neg_b=*/false, b,
      /*c_mod=*/(short)0, c, /*reuse_a=*/false, /*reuse_b=*/false);
}

// Load one operand row-slice in the CDNA5 16-bit A/B lane layout:
// lane half hsel=0 carries K = {kk+0..7, kk+16..23}, hsel=1 carries {kk+8..15, kk+24..31}.
// rowbase points at element (row, kk) of a row-major matrix (global or LDS).
static __device__ inline v16bf load_tile16(const bf16* rowbase, int hsel) {
  bf16x8 lo = *reinterpret_cast<const bf16x8*>(rowbase + 8 * hsel);
  bf16x8 hi = *reinterpret_cast<const bf16x8*>(rowbase + 16 + 8 * hsel);
  return __builtin_shufflevector(lo, hi, 0,1,2,3,4,5,6,7,8,9,10,11,12,13,14,15);
}

// ---------------------------------------------------------------------------
// CDNA5 async global->LDS copy (ASYNCcnt path). Per ISA 10/15.18:
//   global_load_async_to_lds_b128  vdst(LDS byte addr), vaddr(64b global), off
// Generic pointers to __shared__ carry the LDS byte offset in addr[31:0]
// (ISA 10.2 aperture mapping), so the low 32 bits give VDST directly.
// ---------------------------------------------------------------------------
static __device__ inline void async_copy_b128(const bf16* gsrc, const bf16* ldst) {
  unsigned l = (unsigned)(unsigned long long)ldst;
  unsigned long long g = (unsigned long long)gsrc;
  asm volatile("global_load_async_to_lds_b128 %0, %1, off"
               :: "v"(l), "v"(g) : "memory");
}
static __device__ inline void wait_async0() {
  asm volatile("s_wait_asynccnt 0x0" ::: "memory");
}

// ---------------------------------------------------------------------------
// fp32 -> bf16 conversion (grid-stride)
// ---------------------------------------------------------------------------
__global__ void cvt_f32_bf16(const float* __restrict__ in, bf16* __restrict__ out, int n) {
  int i = blockIdx.x * blockDim.x + threadIdx.x;
  int stride = gridDim.x * blockDim.x;
  for (; i < n; i += stride) out[i] = (bf16)in[i];
}

// ---------------------------------------------------------------------------
// Generic bf16 GEMM: Y = X[BS,D] * W[D,D]^T  (+bias) (*scale)
//   mode 0: bf16 row-major out [BS,D]          (Q / K projections)
//   mode 1: bf16 out transposed per head: Vt[b,h,d,s]   (V projection)
//   mode 2: fp32 row-major out [BS,D]          (final output projection)
// Block = 256 threads (8 waves) computes a 128x64 tile. Per 32-wide k-step the
// A slab (128x32) and the block-shared B tile (64x32) are staged in LDS with
// async copies, double-buffered: step s+1's copy overlaps step s's WMMAs.
// ---------------------------------------------------------------------------
__global__ void gemm_bf16(const bf16* __restrict__ X, const bf16* __restrict__ W,
                          const float* __restrict__ bias, int has_bias, float scale,
                          bf16* __restrict__ Yb, float* __restrict__ Yf, int mode) {
  __shared__ bf16 smA[2][128 * 32];   // 8 KB per buffer
  __shared__ bf16 smB[2][64 * 32];    // 4 KB per buffer

  const int tid  = threadIdx.x;
  const int lane = tid & 31;
  const int wave = tid >> 5;
  const int mblk   = blockIdx.x >> 4;        // 0..63
  const int m_base = mblk * 128;
  const int n0     = (blockIdx.x & 15) * 64; // col strip base
  const int hsel = lane >> 4;
  const int l15  = lane & 15;

  // async-copy index split: B tile = 256 x 16B chunks, A slab = 512 x 16B chunks
  const int br = tid >> 2, bc = tid & 3;     // B: row 0..63, chunk 0..3

  auto issue_copies = [&](int buf, int kk) {
    // B tile: W rows n0..n0+63, k slice [kk, kk+32)
    async_copy_b128(W + (size_t)(n0 + br) * DMODEL + kk + bc * 8,
                    &smB[buf][br * 32 + bc * 8]);
    // A slab: X rows m_base..m_base+127, k slice [kk, kk+32)
#pragma unroll
    for (int i = 0; i < 2; i++) {
      int idx = tid + i * 256;
      int ar = idx >> 2, ac = idx & 3;
      async_copy_b128(X + (size_t)(m_base + ar) * DMODEL + kk + ac * 8,
                      &smA[buf][ar * 32 + ac * 8]);
    }
  };

  v8f acc[4] = {};

  issue_copies(0, 0);
  wait_async0();
  __syncthreads();

  constexpr int KSTEPS = DMODEL / 32;   // 32
  for (int s = 0; s < KSTEPS; s++) {
    const int cur = s & 1;
    if (s + 1 < KSTEPS) issue_copies(cur ^ 1, (s + 1) * 32);

    const bf16* arow = &smA[cur][(wave * 16 + l15) * 32];
    const bf16* brow = &smB[cur][l15 * 32];
    v16bf a  = load_tile16(arow, hsel);
    v16bf b0 = load_tile16(brow, hsel);
    v16bf b1 = load_tile16(brow + 16 * 32, hsel);
    v16bf b2 = load_tile16(brow + 32 * 32, hsel);
    v16bf b3 = load_tile16(brow + 48 * 32, hsel);
    acc[0] = wmma_bf16(a, b0, acc[0]);
    acc[1] = wmma_bf16(a, b1, acc[1]);
    acc[2] = wmma_bf16(a, b2, acc[2]);
    acc[3] = wmma_bf16(a, b3, acc[3]);

    wait_async0();      // this wave's copies for buf cur^1 are done
    __syncthreads();    // everyone finished reading buf cur / writing cur^1
  }

#pragma unroll
  for (int t = 0; t < 4; t++) {
    const int col = n0 + t * 16 + l15;
    const float bv = has_bias ? bias[col] : 0.0f;
#pragma unroll
    for (int r = 0; r < 8; r++) {
      const int mrow = m_base + wave * 16 + r + 8 * hsel; // C/D: m = vgpr + 8*half, n = lane&15
      const float v = (acc[t][r] + bv) * scale;
      if (mode == 0) {
        Yb[(size_t)mrow * DMODEL + col] = (bf16)v;
      } else if (mode == 1) {
        const int b  = mrow >> 10, sq = mrow & 1023;
        const int h  = col >> 6,  d  = col & 63;
        Yb[(((size_t)((b * NHEAD + h) * DHEAD + d)) << 10) + sq] = (bf16)v;
      } else {
        Yf[(size_t)mrow * DMODEL + col] = v;
      }
    }
  }
}

// ---------------------------------------------------------------------------
// qk[b,h,q,k] = (Q_bh @ K_bh^T) + mask[q,k]   (Q,K already scaled by dh^-0.25)
// One wave per 16x64 strip; K-loop is only dh/32 = 2 steps, register path.
// ---------------------------------------------------------------------------
__global__ void qk_scores(const bf16* __restrict__ Q, const bf16* __restrict__ K,
                          const float* __restrict__ mask, float* __restrict__ qk) {
  const int lane = threadIdx.x & 31;
  const int wave = threadIdx.x >> 5;
  const int wid  = blockIdx.x * 8 + wave;   // 0 .. 131071
  const int bh   = wid >> 10;               // 0 .. 127
  const int rem  = wid & 1023;
  const int m16  = rem >> 4;                // q tile
  const int n0   = (rem & 15) * 64;         // k strip
  const int b    = bh >> 4, h = bh & 15;
  const int hsel = lane >> 4;
  const int l15  = lane & 15;

  const bf16* qrow = Q + (size_t)(b * SEQ + m16 * 16 + l15) * DMODEL + h * DHEAD;
  const bf16* krow = K + (size_t)(b * SEQ + n0 + l15) * DMODEL + h * DHEAD;

  v8f acc[4] = {};
#pragma unroll
  for (int kk = 0; kk < DHEAD; kk += 32) {
    v16bf a = load_tile16(qrow + kk, hsel);
#pragma unroll
    for (int t = 0; t < 4; t++) {
      v16bf bt = load_tile16(krow + (size_t)(t * 16) * DMODEL + kk, hsel);
      acc[t] = wmma_bf16(a, bt, acc[t]);
    }
  }

  float* out = qk + (size_t)bh * SEQ * SEQ;
#pragma unroll
  for (int t = 0; t < 4; t++) {
    const int kcol = n0 + t * 16 + l15;
#pragma unroll
    for (int r = 0; r < 8; r++) {
      const int q = m16 * 16 + r + 8 * hsel;
      out[(size_t)q * SEQ + kcol] = acc[t][r] + mask[(size_t)q * SEQ + kcol];
    }
  }
}

// ---------------------------------------------------------------------------
// Fused softmax + P@V. One wave per (b,h,16-row q block).
// Pass 1: row max / sum-exp over the materialized qk (fp32, L2-resident).
// Pass 2: build bf16 P tiles on the fly in A-operand layout, WMMA against Vt.
// ---------------------------------------------------------------------------
__global__ void softmax_wv(const float* __restrict__ qk, const bf16* __restrict__ Vt,
                           bf16* __restrict__ WVb) {
  const int lane = threadIdx.x & 31;
  const int wave = threadIdx.x >> 5;
  const int wid  = blockIdx.x * 8 + wave;   // 0 .. 8191
  const int bh   = wid >> 6;                // 0 .. 127
  const int m16  = wid & 63;
  const int b    = bh >> 4, h = bh & 15;
  const int hsel = lane >> 4;
  const int l15  = lane & 15;

  // --- pass 1: stats for row (m16*16 + l15); lane halves split the 1024 cols ---
  const float* rowp = qk + ((size_t)bh * SEQ + m16 * 16 + l15) * SEQ + hsel * 512;
  float mx = -3.0e38f;
  for (int j = 0; j < 128; j++) {
    floatx4 v = reinterpret_cast<const floatx4*>(rowp)[j];
    mx = fmaxf(mx, fmaxf(fmaxf(v[0], v[1]), fmaxf(v[2], v[3])));
  }
  mx = fmaxf(mx, __shfl_xor(mx, 16, 32));
  float sm = 0.0f;
  for (int j = 0; j < 128; j++) {
    floatx4 v = reinterpret_cast<const floatx4*>(rowp)[j];
    sm += __expf(v[0] - mx) + __expf(v[1] - mx) + __expf(v[2] - mx) + __expf(v[3] - mx);
  }
  sm += __shfl_xor(sm, 16, 32);
  const float inv = 1.0f / sm;

  // --- pass 2: P @ V over K = SEQ ---
  const float* arow = qk + ((size_t)bh * SEQ + m16 * 16 + l15) * SEQ;
  const bf16* vrow  = Vt + ((size_t)bh * DHEAD + l15) * SEQ;

  v8f acc[4] = {};
  for (int kk = 0; kk < SEQ; kk += 32) {
    const float* ab = arow + kk + 8 * hsel;
    floatx4 x0 = *reinterpret_cast<const floatx4*>(ab);
    floatx4 x1 = *reinterpret_cast<const floatx4*>(ab + 4);
    floatx4 x2 = *reinterpret_cast<const floatx4*>(ab + 16);
    floatx4 x3 = *reinterpret_cast<const floatx4*>(ab + 20);
    float tv[16];
#pragma unroll
    for (int j = 0; j < 4; j++) {
      tv[j] = x0[j]; tv[4 + j] = x1[j]; tv[8 + j] = x2[j]; tv[12 + j] = x3[j];
    }
    v16bf a;
#pragma unroll
    for (int j = 0; j < 16; j++) a[j] = (bf16)(__expf(tv[j] - mx) * inv);

#pragma unroll
    for (int t = 0; t < 4; t++) {
      v16bf bt = load_tile16(vrow + (size_t)(t * 16) * SEQ + kk, hsel);
      acc[t] = wmma_bf16(a, bt, acc[t]);
    }
  }

#pragma unroll
  for (int t = 0; t < 4; t++) {
    const int col = h * DHEAD + t * 16 + l15;
#pragma unroll
    for (int r = 0; r < 8; r++) {
      const int s = m16 * 16 + r + 8 * hsel;
      WVb[(size_t)(b * SEQ + s) * DMODEL + col] = (bf16)acc[t][r];
    }
  }
}

// ---------------------------------------------------------------------------
// Host-side orchestration
// ---------------------------------------------------------------------------
extern "C" void kernel_launch(void* const* d_in, const int* in_sizes, int n_in,
                              void* d_out, int out_size, void* d_ws, size_t ws_size,
                              hipStream_t stream) {
  (void)in_sizes; (void)n_in; (void)out_size; (void)ws_size;

  const float* x    = (const float*)d_in[0];
  const float* mask = (const float*)d_in[1];
  const float* Wq   = (const float*)d_in[2];
  const float* bq   = (const float*)d_in[3];
  const float* Wk   = (const float*)d_in[4];
  const float* Wv   = (const float*)d_in[5];
  const float* bv   = (const float*)d_in[6];
  const float* Wo   = (const float*)d_in[7];
  const float* bo   = (const float*)d_in[8];

  float* outp = (float*)d_out;                       // [B,S,D]
  float* qkp  = outp + (size_t)BS * DMODEL;          // [B,H,S,S]

  const size_t MB = 1u << 20;
  char* ws = (char*)d_ws;
  bf16* xb  = (bf16*)(ws + 0 * MB);    // [BS,D]
  bf16* qb  = (bf16*)(ws + 16 * MB);   // [BS,D]  scaled Q
  bf16* kb  = (bf16*)(ws + 32 * MB);   // [BS,D]  scaled K
  bf16* vt  = (bf16*)(ws + 48 * MB);   // [B,H,dh,S]  V transposed
  bf16* wvb = (bf16*)(ws + 64 * MB);   // [BS,D]  attention output
  bf16* wqb = (bf16*)(ws + 80 * MB);
  bf16* wkb = (bf16*)(ws + 82 * MB);
  bf16* wvw = (bf16*)(ws + 84 * MB);
  bf16* wob = (bf16*)(ws + 86 * MB);

  const float scale = 0.35355339059327373f;  // 64^-0.25

  // 1) fp32 -> bf16
  cvt_f32_bf16<<<4096, 256, 0, stream>>>(x,  xb,  BS * DMODEL);
  cvt_f32_bf16<<<2048, 256, 0, stream>>>(Wq, wqb, DMODEL * DMODEL);
  cvt_f32_bf16<<<2048, 256, 0, stream>>>(Wk, wkb, DMODEL * DMODEL);
  cvt_f32_bf16<<<2048, 256, 0, stream>>>(Wv, wvw, DMODEL * DMODEL);
  cvt_f32_bf16<<<2048, 256, 0, stream>>>(Wo, wob, DMODEL * DMODEL);

  // 2) projections: 64 row-blocks x 16 col strips -> 1024 blocks x 256 thr
  gemm_bf16<<<1024, 256, 0, stream>>>(xb, wqb, bq, 1, scale, qb,  nullptr, 0); // Q
  gemm_bf16<<<1024, 256, 0, stream>>>(xb, wkb, bq, 0, scale, kb,  nullptr, 0); // K (no bias)
  gemm_bf16<<<1024, 256, 0, stream>>>(xb, wvw, bv, 1, 1.0f,  vt,  nullptr, 1); // V -> Vt

  // 3) scores + mask -> qk output (131072 waves)
  qk_scores<<<16384, 256, 0, stream>>>(qb, kb, mask, qkp);

  // 4) softmax + P@V (8192 waves)
  softmax_wv<<<1024, 256, 0, stream>>>(qkp, vt, wvb);

  // 5) final projection: out = wv @ Wo^T + bo (fp32)
  gemm_bf16<<<1024, 256, 0, stream>>>(wvb, wob, bo, 1, 1.0f, nullptr, outp, 2);
}
